// VAE_4939212390535
// MI455X (gfx1250) — matrix operand order repaired and checked by reading, not compiled, not run on hardware
//
#include <hip/hip_runtime.h>
#include <cstdint>
#include <cstddef>

// ---------------- problem dims ----------------
#define T_STEPS 64
#define RDIM    130
#define HDIM    1024
#define ZDIM    128
#define BATCH   256
#define KX      160   // R padded to mult of 32
#define KC0     288   // (R+Z)=258 padded to mult of 32
#define G3H     3072  // 3*H
#define NOUT    192   // Wout rows padded to mult of 64
#define LDSW    20    // LDS row stride in u32 (16 data + 4 pad)

typedef __attribute__((ext_vector_type(16))) __bf16 v16bf;
typedef __attribute__((ext_vector_type(8)))  float  v8f;
typedef unsigned int u32x4 __attribute__((ext_vector_type(4)));
typedef int          i32x8 __attribute__((ext_vector_type(8)));
typedef int          i32x4 __attribute__((ext_vector_type(4)));

#if defined(__HIP_DEVICE_COMPILE__) && __has_builtin(__builtin_amdgcn_tensor_load_to_lds) && __has_builtin(__builtin_amdgcn_s_wait_tensorcnt)
#define USE_TDM 1
#else
#define USE_TDM 0
#endif

__device__ __forceinline__ uint16_t f2bf(float f) {
  uint32_t u = __float_as_uint(f);
  u += 0x7FFFu + ((u >> 16) & 1u);   // round-to-nearest-even
  return (uint16_t)(u >> 16);
}
__device__ __forceinline__ float sigmoidf_(float x) { return 1.0f / (1.0f + expf(-x)); }

// ---------------- conversion / init kernels ----------------
__global__ __launch_bounds__(256)
void k_cvt_pad_bf16(const float* __restrict__ src, uint16_t* __restrict__ dst,
                    int srows, int drows, int scols, int dcols) {
  int i = blockIdx.x * blockDim.x + threadIdx.x;
  int tot = drows * dcols;
  if (i >= tot) return;
  int r = i / dcols, c = i - r * dcols;
  float v = (r < srows && c < scols) ? src[(size_t)r * scols + c] : 0.0f;
  dst[i] = f2bf(v);
}

__global__ __launch_bounds__(256)
void k_zero_u32(uint32_t* __restrict__ p, int n) {
  int i = blockIdx.x * blockDim.x + threadIdx.x;
  if (i < n) p[i] = 0u;
}

__global__ __launch_bounds__(256)
void k_concat_bf(const float* __restrict__ hf, const float* __restrict__ hb,
                 uint16_t* __restrict__ enc) {
  int i = blockIdx.x * blockDim.x + threadIdx.x;  // B*2H
  if (i >= BATCH * 2 * HDIM) return;
  int b = i >> 11, j = i & 2047;
  float v = (j < HDIM) ? hf[(size_t)b * HDIM + j] : hb[(size_t)b * HDIM + (j - HDIM)];
  enc[i] = f2bf(v);
}

__global__ __launch_bounds__(256)
void k_decinp_init(const uint16_t* __restrict__ zbf, uint16_t* __restrict__ inp) {
  int i = blockIdx.x * blockDim.x + threadIdx.x;  // B*KC0
  if (i >= BATCH * KC0) return;
  int b = i / KC0, c = i - b * KC0;
  uint16_t v = 0;
  if (c == RDIM - 1)                     v = 0x3F80u;   // bf16 1.0
  else if (c >= RDIM && c < RDIM + ZDIM) v = zbf[(size_t)b * ZDIM + (c - RDIM)];
  inp[i] = v;
}

// ---------------- TDM staging of a 64xN-row, 32-K weight slab into LDS ----------------
#if USE_TDM
__device__ __forceinline__ void tdm_stage(const uint16_t* W, int ldw, int n0, int k0,
                                          uint32_t lds_addr, int K) {
  uint64_t ga = (uint64_t)(uintptr_t)W + ((uint64_t)(uint32_t)(n0) * (uint32_t)ldw + (uint32_t)k0) * 2ull;
  u32x4 g0;
  g0[0] = 1u;                                   // count=1, user descriptor
  g0[1] = lds_addr;                             // LDS dest (bytes)
  g0[2] = (uint32_t)ga;                         // global_addr[31:0]
  g0[3] = ((uint32_t)(ga >> 32) & 0x01FFFFFFu)  // global_addr[56:32]
          | (2u << 30);                         // type=2 ("image")
  i32x8 g1;
  g1[0] = (1 << 16)        // data_size = 2 bytes
        | (1 << 20)        // pad_enable
        | (3 << 22)        // pad_interval: 16 DWORDs
        | (3 << 25);       // pad_amount:   4 DWORDs  -> LDS row stride = 20 u32
  g1[1] = (K & 0xFFFF) << 16;                     // tensor_dim0[15:0]
  g1[2] = ((K >> 16) & 0xFFFF) | (64 << 16);      // tensor_dim0[31:16] | tensor_dim1[15:0]=64
  g1[3] = (32 << 16);                             // tensor_dim1 hi=0 | tile_dim0=32
  g1[4] = 64;                                     // tile_dim1=64, tile_dim2=0
  g1[5] = ldw;                                    // tensor_dim0_stride[31:0]
  g1[6] = 0;                                      // stride hi | dim1_stride lo
  g1[7] = 0;
  i32x4 gz; gz[0] = 0; gz[1] = 0; gz[2] = 0; gz[3] = 0;
#if defined(__clang_major__) && (__clang_major__ >= 23)
  i32x8 gz8;
  gz8[0] = 0; gz8[1] = 0; gz8[2] = 0; gz8[3] = 0;
  gz8[4] = 0; gz8[5] = 0; gz8[6] = 0; gz8[7] = 0;
  __builtin_amdgcn_tensor_load_to_lds(g0, g1, gz, gz, gz8, 0);
#else
  __builtin_amdgcn_tensor_load_to_lds(g0, g1, gz, gz, 0);
#endif
}
#endif

// ---------------- WMMA GEMM: C[B,Npad] = A[B,K] * W[Npad,K]^T (+bias, act) ----------------
// block = 128 threads (4 waves); block tile = 128(M) x 64(N); wave tile = 32(M) x 64(N).
// Weight slab (64 rows x 32 K) staged in LDS via TDM (double-buffered), shared by all waves.
// Requires: N multiple of 64 (weights row-padded), K multiple of 32, M = 256.
// mode: 0 = identity, 1 = tanh, 2 = exp(0.5*x)
__global__ __launch_bounds__(128)
void k_gemm(const uint16_t* __restrict__ A, int lda,
            const uint16_t* __restrict__ W, int ldw,
            const float* __restrict__ bias,
            float* __restrict__ C, int ldc,
            uint16_t* __restrict__ Cbf,
            int K, int mode) {
  __shared__ uint32_t Bt[2][64 * LDSW];

  const int lane = threadIdx.x & 31;
  const int wv   = threadIdx.x >> 5;
  const int n0   = blockIdx.x * 64;
  const int m0   = blockIdx.y * 128;
  const int ml   = lane & 15;
  const int hi   = lane >> 4;
  const int mrow0 = m0 + wv * 32 + ml;   // A fragment 0 rows
  const int mrow1 = mrow0 + 16;          // A fragment 1 rows

  const v8f vzero = {0.f, 0.f, 0.f, 0.f, 0.f, 0.f, 0.f, 0.f};
  v8f acc[8];
#pragma unroll
  for (int i = 0; i < 8; ++i) acc[i] = vzero;

#if USE_TDM
  const uint32_t lds0 = (uint32_t)(uintptr_t)(&Bt[0][0]);
  const uint32_t lds1 = (uint32_t)(uintptr_t)(&Bt[1][0]);
  if (wv == 0) {
    tdm_stage(W, ldw, n0, 0, lds0, K);
    __builtin_amdgcn_s_wait_tensorcnt((short)0);
  }
  __syncthreads();
#else
  const int srow  = threadIdx.x >> 1;     // staging: 2 threads per n-row
  const int shalf = threadIdx.x & 1;
  const uint16_t* wsrc = W + (size_t)(n0 + srow) * ldw + shalf * 16;
  uint32_t* sdst = &Bt[0][srow * LDSW + shalf * 8];
#endif

  int cur = 0;
  for (int k0 = 0; k0 < K; k0 += 32) {
    const bool more = (k0 + 32) < K;
#if USE_TDM
    if (more && wv == 0) tdm_stage(W, ldw, n0, k0 + 32, cur ? lds0 : lds1, K);
#else
    {
      uint4 w0 = *(const uint4*)(wsrc + k0);
      uint4 w1 = *(const uint4*)(wsrc + k0 + 8);
      *(uint4*)(sdst)     = w0;
      *(uint4*)(sdst + 4) = w1;
      __syncthreads();
    }
#endif
    const uint32_t* Bc = &Bt[cur][0];

    // A fragments (16x32 bf16 each), ISA per-lane K swizzle -> 2x b128 per frag
    union { uint32_t u[8]; v16bf v; } a0, a1;
    const uint16_t* Ap0 = A + (size_t)mrow0 * lda + k0;
    const uint16_t* Ap1 = A + (size_t)mrow1 * lda + k0;
#pragma unroll
    for (int j = 0; j < 4; ++j) {
      a0.u[j]     = *(const uint32_t*)(Ap0 + 2 * j + 8 * hi);
      a0.u[4 + j] = *(const uint32_t*)(Ap0 + 16 + 2 * j + 8 * hi);
      a1.u[j]     = *(const uint32_t*)(Ap1 + 2 * j + 8 * hi);
      a1.u[4 + j] = *(const uint32_t*)(Ap1 + 16 + 2 * j + 8 * hi);
    }

#pragma unroll
    for (int nn = 0; nn < 4; ++nn) {
      union { uint32_t u[8]; v16bf v; } b;
#pragma unroll
      for (int j = 0; j < 8; ++j) b.u[j] = Bc[(nn * 16 + ml) * LDSW + hi * 4 + j + (j & 4)];
      acc[nn]     = __builtin_amdgcn_wmma_f32_16x16x32_bf16(
          false, a0.v, false, b.v, (short)0, acc[nn], false, false);
      acc[4 + nn] = __builtin_amdgcn_wmma_f32_16x16x32_bf16(
          false, a1.v, false, b.v, (short)0, acc[4 + nn], false, false);
    }

    if (more) {
#if USE_TDM
      if (wv == 0) __builtin_amdgcn_s_wait_tensorcnt((short)0);
      __syncthreads();
      cur ^= 1;
#else
      __syncthreads();
#endif
    }
  }

  // store: frag (g,nn), VGPR r -> row = m0 + 32*wv + 16*g + 8*hi + r, col = n0 + 16*nn + ml
#pragma unroll
  for (int g = 0; g < 2; ++g) {
#pragma unroll
    for (int nn = 0; nn < 4; ++nn) {
      int col = n0 + nn * 16 + ml;
      float bi = bias ? bias[col] : 0.0f;
      union { v8f v; float f[8]; } t; t.v = acc[g * 4 + nn];
#pragma unroll
      for (int r = 0; r < 8; ++r) {
        int row = m0 + wv * 32 + g * 16 + hi * 8 + r;
        float v = t.f[r] + bi;
        if (mode == 1)      v = tanhf(v);
        else if (mode == 2) v = expf(0.5f * v);
        C[(size_t)row * ldc + col] = v;
        if (Cbf) Cbf[(size_t)row * ldc + col] = f2bf(v);
      }
    }
  }
}

// ---------------- fused GRU pointwise ----------------
__global__ __launch_bounds__(256)
void k_gru_pw(const float* __restrict__ gi, const float* __restrict__ gh,
              const float* __restrict__ hprev,
              float* __restrict__ h, uint16_t* __restrict__ hbf) {
  int i = blockIdx.x * blockDim.x + threadIdx.x;  // B*H
  if (i >= BATCH * HDIM) return;
  int b = i >> 10, j = i & 1023;
  const float* gib = gi + (size_t)b * G3H;
  const float* ghb = gh + (size_t)b * G3H;
  float r  = sigmoidf_(gib[j]            + ghb[j]);
  float zg = sigmoidf_(gib[HDIM + j]     + ghb[HDIM + j]);
  float n  = tanhf    (gib[2 * HDIM + j] + r * ghb[2 * HDIM + j]);
  float o  = (1.0f - zg) * n + zg * hprev[i];
  h[i]   = o;
  hbf[i] = f2bf(o);
}

// ---------------- log_softmax + argmax + next one-hot ----------------
__global__ __launch_bounds__(128)
void k_softmax_argmax(const float* __restrict__ logits, float* __restrict__ outp,
                      int t, uint16_t* __restrict__ inp) {
  __shared__ float smax[128];
  __shared__ int   sidx[128];
  __shared__ float ssum[128];
  int b = blockIdx.x, tid = threadIdx.x;
  const float* lg = logits + (size_t)b * NOUT;

  float mx = -INFINITY; int mi = 0;
  for (int c = tid; c < RDIM; c += 128) {
    float v = lg[c];
    if (v > mx) { mx = v; mi = c; }
  }
  smax[tid] = mx; sidx[tid] = mi;
  __syncthreads();
  for (int s = 64; s > 0; s >>= 1) {
    if (tid < s) {
      float o = smax[tid + s]; int oi = sidx[tid + s];
      if (o > smax[tid] || (o == smax[tid] && oi < sidx[tid])) { smax[tid] = o; sidx[tid] = oi; }
    }
    __syncthreads();
  }
  float gmax = smax[0]; int gidx = sidx[0];

  float acc = 0.0f;
  for (int c = tid; c < RDIM; c += 128) acc += expf(lg[c] - gmax);
  ssum[tid] = acc;
  __syncthreads();
  for (int s = 64; s > 0; s >>= 1) {
    if (tid < s) ssum[tid] += ssum[tid + s];
    __syncthreads();
  }
  float lse = logf(ssum[0]) + gmax;

  float* dst = outp + ((size_t)b * T_STEPS + t) * RDIM;
  for (int c = tid; c < RDIM; c += 128) dst[c] = lg[c] - lse;
  for (int c = tid; c < RDIM; c += 128)
    inp[(size_t)b * KC0 + c] = (c == gidx) ? (uint16_t)0x3F80u : (uint16_t)0u;
}

// ---------------- host orchestration ----------------
extern "C" void kernel_launch(void* const* d_in, const int* in_sizes, int n_in,
                              void* d_out, int out_size, void* d_ws, size_t ws_size,
                              hipStream_t stream) {
  (void)in_sizes; (void)n_in; (void)out_size; (void)ws_size;
  const float* x      = (const float*)d_in[0];
  const float* Wih_f  = (const float*)d_in[1];
  const float* Whh_f  = (const float*)d_in[2];
  const float* bih_f  = (const float*)d_in[3];
  const float* bhh_f  = (const float*)d_in[4];
  const float* Wih_b  = (const float*)d_in[5];
  const float* Whh_b  = (const float*)d_in[6];
  const float* bih_b  = (const float*)d_in[7];
  const float* bhh_b  = (const float*)d_in[8];
  const float* Wmu    = (const float*)d_in[9];
  const float* bmu    = (const float*)d_in[10];
  const float* Wvar   = (const float*)d_in[11];
  const float* bvar   = (const float*)d_in[12];
  const float* Wc0_ih = (const float*)d_in[13];
  const float* Wc0_hh = (const float*)d_in[14];
  const float* bc0_ih = (const float*)d_in[15];
  const float* bc0_hh = (const float*)d_in[16];
  const float* Wc1_ih = (const float*)d_in[17];
  const float* Wc1_hh = (const float*)d_in[18];
  const float* bc1_ih = (const float*)d_in[19];
  const float* bc1_hh = (const float*)d_in[20];
  const float* Wc2_ih = (const float*)d_in[21];
  const float* Wc2_hh = (const float*)d_in[22];
  const float* bc2_ih = (const float*)d_in[23];
  const float* bc2_hh = (const float*)d_in[24];
  const float* Wout   = (const float*)d_in[25];
  const float* bout   = (const float*)d_in[26];
  const float* Winit  = (const float*)d_in[27];
  const float* binit  = (const float*)d_in[28];
  float* out = (float*)d_out;

  char* base = (char*)d_ws;
  size_t off = 0;
  auto alloc = [&](size_t bytes) -> void* {
    off = (off + 255) & ~(size_t)255;
    void* p = base + off;
    off += bytes;
    return p;
  };

  // bf16 copies (weights padded along K; Wout padded along N too)
  uint16_t* xbf       = (uint16_t*)alloc((size_t)BATCH * T_STEPS * KX * 2);
  uint16_t* Wih_f_bf  = (uint16_t*)alloc((size_t)G3H * KX * 2);
  uint16_t* Wih_b_bf  = (uint16_t*)alloc((size_t)G3H * KX * 2);
  uint16_t* Whh_f_bf  = (uint16_t*)alloc((size_t)G3H * HDIM * 2);
  uint16_t* Whh_b_bf  = (uint16_t*)alloc((size_t)G3H * HDIM * 2);
  uint16_t* Wmu_bf    = (uint16_t*)alloc((size_t)ZDIM * 2 * HDIM * 2);
  uint16_t* Wvar_bf   = (uint16_t*)alloc((size_t)ZDIM * 2 * HDIM * 2);
  uint16_t* Wc0ih_bf  = (uint16_t*)alloc((size_t)G3H * KC0 * 2);
  uint16_t* Wc0hh_bf  = (uint16_t*)alloc((size_t)G3H * HDIM * 2);
  uint16_t* Wc1ih_bf  = (uint16_t*)alloc((size_t)G3H * HDIM * 2);
  uint16_t* Wc1hh_bf  = (uint16_t*)alloc((size_t)G3H * HDIM * 2);
  uint16_t* Wc2ih_bf  = (uint16_t*)alloc((size_t)G3H * HDIM * 2);
  uint16_t* Wc2hh_bf  = (uint16_t*)alloc((size_t)G3H * HDIM * 2);
  uint16_t* Wout_bf   = (uint16_t*)alloc((size_t)NOUT * HDIM * 2);
  uint16_t* Winit_bf  = (uint16_t*)alloc((size_t)HDIM * ZDIM * 2);

  // activations
  float*    gi     = (float*)alloc((size_t)BATCH * G3H * 4);
  float*    gh     = (float*)alloc((size_t)BATCH * G3H * 4);
  float*    hf     = (float*)alloc((size_t)BATCH * HDIM * 4);
  float*    hb     = (float*)alloc((size_t)BATCH * HDIM * 4);
  float*    h0     = (float*)alloc((size_t)BATCH * HDIM * 4);
  float*    h1     = (float*)alloc((size_t)BATCH * HDIM * 4);
  float*    h2     = (float*)alloc((size_t)BATCH * HDIM * 4);
  float*    logits = (float*)alloc((size_t)BATCH * NOUT * 4);
  uint16_t* hf_bf  = (uint16_t*)alloc((size_t)BATCH * HDIM * 2);
  uint16_t* hb_bf  = (uint16_t*)alloc((size_t)BATCH * HDIM * 2);
  uint16_t* h0_bf  = (uint16_t*)alloc((size_t)BATCH * HDIM * 2);
  uint16_t* h1_bf  = (uint16_t*)alloc((size_t)BATCH * HDIM * 2);
  uint16_t* h2_bf  = (uint16_t*)alloc((size_t)BATCH * HDIM * 2);
  uint16_t* enc_bf = (uint16_t*)alloc((size_t)BATCH * 2 * HDIM * 2);
  uint16_t* z_bf   = (uint16_t*)alloc((size_t)BATCH * ZDIM * 2);
  uint16_t* decinp = (uint16_t*)alloc((size_t)BATCH * KC0 * 2);

  auto cvt = [&](const float* s, uint16_t* d, int sr, int dr, int sc, int dc) {
    int tot = dr * dc;
    k_cvt_pad_bf16<<<(tot + 255) / 256, 256, 0, stream>>>(s, d, sr, dr, sc, dc);
  };
  auto gemm = [&](const uint16_t* A, int lda, const uint16_t* Wp, int ldw,
                  const float* bias, float* C, int ldc, uint16_t* Cbf,
                  int N, int K, int mode) {
    dim3 grid(N / 64, BATCH / 128);
    k_gemm<<<grid, 128, 0, stream>>>(A, lda, Wp, ldw, bias, C, ldc, Cbf, K, mode);
  };
  auto gru = [&](const float* gip, const float* ghp, const float* hprev,
                 float* h, uint16_t* hbf) {
    k_gru_pw<<<(BATCH * HDIM) / 256, 256, 0, stream>>>(gip, ghp, hprev, h, hbf);
  };

  // ---- phase 0: convert weights + input to bf16 ----
  cvt(x,      xbf,      BATCH * T_STEPS, BATCH * T_STEPS, RDIM, KX);
  cvt(Wih_f,  Wih_f_bf, G3H, G3H, RDIM, KX);
  cvt(Wih_b,  Wih_b_bf, G3H, G3H, RDIM, KX);
  cvt(Whh_f,  Whh_f_bf, G3H, G3H, HDIM, HDIM);
  cvt(Whh_b,  Whh_b_bf, G3H, G3H, HDIM, HDIM);
  cvt(Wmu,    Wmu_bf,   ZDIM, ZDIM, 2 * HDIM, 2 * HDIM);
  cvt(Wvar,   Wvar_bf,  ZDIM, ZDIM, 2 * HDIM, 2 * HDIM);
  cvt(Wc0_ih, Wc0ih_bf, G3H, G3H, RDIM + ZDIM, KC0);
  cvt(Wc0_hh, Wc0hh_bf, G3H, G3H, HDIM, HDIM);
  cvt(Wc1_ih, Wc1ih_bf, G3H, G3H, HDIM, HDIM);
  cvt(Wc1_hh, Wc1hh_bf, G3H, G3H, HDIM, HDIM);
  cvt(Wc2_ih, Wc2ih_bf, G3H, G3H, HDIM, HDIM);
  cvt(Wc2_hh, Wc2hh_bf, G3H, G3H, HDIM, HDIM);
  cvt(Wout,   Wout_bf,  RDIM, NOUT, HDIM, HDIM);   // pad rows 130->192 with zeros
  cvt(Winit,  Winit_bf, HDIM, HDIM, ZDIM, ZDIM);

  // zero encoder hidden state (f32 + bf16)
  k_zero_u32<<<(BATCH * HDIM) / 256, 256, 0, stream>>>((uint32_t*)hf, BATCH * HDIM);
  k_zero_u32<<<(BATCH * HDIM) / 256, 256, 0, stream>>>((uint32_t*)hb, BATCH * HDIM);
  k_zero_u32<<<(BATCH * HDIM / 2) / 256, 256, 0, stream>>>((uint32_t*)hf_bf, BATCH * HDIM / 2);
  k_zero_u32<<<(BATCH * HDIM / 2) / 256, 256, 0, stream>>>((uint32_t*)hb_bf, BATCH * HDIM / 2);

  // ---- phase 1: bidirectional encoder GRU ----
  const int ldx = T_STEPS * KX;  // row stride of x in bf16 time-sliced view
  for (int t = 0; t < T_STEPS; ++t) {
    gemm(xbf + (size_t)t * KX, ldx, Wih_f_bf, KX,  bih_f, gi, G3H, nullptr, G3H, KX, 0);
    gemm(hf_bf, HDIM,          Whh_f_bf, HDIM, bhh_f, gh, G3H, nullptr, G3H, HDIM, 0);
    gru(gi, gh, hf, hf, hf_bf);
    int tb = T_STEPS - 1 - t;
    gemm(xbf + (size_t)tb * KX, ldx, Wih_b_bf, KX,  bih_b, gi, G3H, nullptr, G3H, KX, 0);
    gemm(hb_bf, HDIM,           Whh_b_bf, HDIM, bhh_b, gh, G3H, nullptr, G3H, HDIM, 0);
    gru(gi, gh, hb, hb, hb_bf);
  }

  // ---- phase 2: latent head ----
  const size_t RECON = (size_t)BATCH * T_STEPS * RDIM;
  k_concat_bf<<<(BATCH * 2 * HDIM) / 256, 256, 0, stream>>>(hf, hb, enc_bf);
  gemm(enc_bf, 2 * HDIM, Wmu_bf,  2 * HDIM, bmu,  out + RECON,                 ZDIM, z_bf,    ZDIM, 2 * HDIM, 0);
  gemm(enc_bf, 2 * HDIM, Wvar_bf, 2 * HDIM, bvar, out + RECON + BATCH * ZDIM, ZDIM, nullptr, ZDIM, 2 * HDIM, 2);
  gemm(z_bf, ZDIM, Winit_bf, ZDIM, binit, h0, HDIM, h0_bf, HDIM, ZDIM, 1);
  k_decinp_init<<<(BATCH * KC0 + 255) / 256, 256, 0, stream>>>(z_bf, decinp);

  // ---- phase 3: greedy decoder, 3 stacked GRU cells ----
  for (int t = 0; t < T_STEPS; ++t) {
    gemm(decinp, KC0, Wc0ih_bf, KC0,  bc0_ih, gi, G3H, nullptr, G3H, KC0, 0);
    gemm(h0_bf, HDIM, Wc0hh_bf, HDIM, bc0_hh, gh, G3H, nullptr, G3H, HDIM, 0);
    gru(gi, gh, h0, h0, h0_bf);
    const uint16_t* h1src_bf = (t == 0) ? h0_bf : h1_bf;
    const float*    h1src    = (t == 0) ? h0    : h1;
    gemm(h0_bf,    HDIM, Wc1ih_bf, HDIM, bc1_ih, gi, G3H, nullptr, G3H, HDIM, 0);
    gemm(h1src_bf, HDIM, Wc1hh_bf, HDIM, bc1_hh, gh, G3H, nullptr, G3H, HDIM, 0);
    gru(gi, gh, h1src, h1, h1_bf);
    const uint16_t* h2src_bf = (t == 0) ? h1_bf : h2_bf;
    const float*    h2src    = (t == 0) ? h1    : h2;
    gemm(h1_bf,    HDIM, Wc2ih_bf, HDIM, bc2_ih, gi, G3H, nullptr, G3H, HDIM, 0);
    gemm(h2src_bf, HDIM, Wc2hh_bf, HDIM, bc2_hh, gh, G3H, nullptr, G3H, HDIM, 0);
    gru(gi, gh, h2src, h2, h2_bf);
    gemm(h2_bf, HDIM, Wout_bf, HDIM, bout, logits, NOUT, nullptr, NOUT, HDIM, 0);
    k_softmax_argmax<<<BATCH, 128, 0, stream>>>(logits, out, t, decinp);
  }
}
